// colorableGNN_17016660427423
// MI455X (gfx1250) — matrix operand order repaired and checked by reading, not compiled, not run on hardware
//
#include <hip/hip_runtime.h>

typedef __attribute__((ext_vector_type(16))) _Float16 v16h;
typedef __attribute__((ext_vector_type(8)))  _Float16 v8h;
typedef __attribute__((ext_vector_type(4)))  _Float16 v4h;
typedef __attribute__((ext_vector_type(8)))  float    v8f;

#define N_NODES  50000
#define N_EDGES  800000
#define N_GRAPHS 64
#define CH       128

union Frag16 { v16h v; v8h h[2]; };

// ---------------- utility kernels ----------------

__global__ void k_zero(float* __restrict__ p, int n) {
    int i = blockIdx.x * blockDim.x + threadIdx.x;
    if (i < n) p[i] = 0.0f;
}

// deg[j] += 1 for each edge dst (float accumulate, converted to rsqrt later)
__global__ void k_degree(const int* __restrict__ col, float* __restrict__ deg, int nE) {
    int i = blockIdx.x * blockDim.x + threadIdx.x;
    if (i < nE) atomicAdd(&deg[col[i]], 1.0f);
}

__global__ void k_invsqrt(float* __restrict__ deg, int n) {
    int i = blockIdx.x * blockDim.x + threadIdx.x;
    if (i < n) deg[i] = rsqrtf(deg[i] + 1.0f);   // self-loop: deg = indeg + 1
}

__global__ void k_edge_norm(const int* __restrict__ row, const int* __restrict__ col,
                            const float* __restrict__ inv, float* __restrict__ enorm, int nE) {
    int i = blockIdx.x * blockDim.x + threadIdx.x;
    if (i < nE) enorm[i] = inv[row[i]] * inv[col[i]];
}

// Convert f32 features -> f16 row-major
__global__ void k_cvt_f16(const float* __restrict__ src, _Float16* __restrict__ dst, int n4) {
    int i = blockIdx.x * blockDim.x + threadIdx.x;
    if (i >= n4) return;
    float4 v = ((const float4*)src)[i];
    v4h o; o[0] = (_Float16)v.x; o[1] = (_Float16)v.y; o[2] = (_Float16)v.z; o[3] = (_Float16)v.w;
    ((v4h*)dst)[i] = o;
}

// Pack a 128x128 f32 weight (row = K/fan_in, col = N/fan_out) into WMMA B-fragment order:
// index = ((nt*4 + ks)*32 + lane)*16 + e ; K = ks*32 + (lane>>4)*16 + e ; N = nt*16 + (lane&15)
__global__ void k_pack_w(const float* __restrict__ W, _Float16* __restrict__ wp) {
    int tid = blockIdx.x * blockDim.x + threadIdx.x;      // 16384 threads
    if (tid >= 8 * 4 * 32 * 16) return;
    int e    = tid & 15;
    int lane = (tid >> 4) & 31;
    int ks   = (tid >> 9) & 3;
    int nt   = tid >> 11;
    int K = ks * 32 + (lane >> 4) * 16 + e;
    int N = nt * 16 + (lane & 15);
    wp[tid] = (_Float16)W[K * CH + N];
}

// ---------------- WMMA GEMM: t[N,128] = h16[N,128] @ Wpacked ----------------
// One wave per 16-row stripe; sweeps 8 column tiles; 4 WMMAs (K=32 each) per tile.
__global__ void k_gemm_wmma(const _Float16* __restrict__ h16,
                            const _Float16* __restrict__ wpack,
                            float* __restrict__ t, int nRows) {
    int wave = (blockIdx.x * blockDim.x + threadIdx.x) >> 5;
    int lane = threadIdx.x & 31;
    int row0 = wave * 16;
    if (row0 >= nRows) return;                 // wave-uniform; EXEC stays all-1s for WMMA
    int m = lane & 15, half = lane >> 4;

    const _Float16* hrow = h16 + (size_t)(row0 + m) * CH;
    Frag16 a[4];
#pragma unroll
    for (int ks = 0; ks < 4; ++ks) {
        int kb = ks * 32;
        // contiguous k order chosen here (not the ISA table order): lanes 0-15 hold K = 0..15; lanes 16-31 hold K = 16..31 (matches packed B)
        a[ks].h[0] = *(const v8h*)(hrow + kb + 16 * half);
        a[ks].h[1] = *(const v8h*)(hrow + kb + 16 * half + 8);
    }

#pragma unroll
    for (int nt = 0; nt < 8; ++nt) {
        v8f acc = {};
#pragma unroll
        for (int ks = 0; ks < 4; ++ks) {
            v16h b = *(const v16h*)(wpack + (size_t)(((nt * 4 + ks) * 32 + lane)) * 16);
            acc = __builtin_amdgcn_wmma_f32_16x16x32_f16(
                    false, a[ks].v, false, b, (short)0, acc, false, false);
        }
        // D layout: VGPR r -> row M = r + 8*half, col N = nt*16 + m
        float* tcol = t + (size_t)row0 * CH + nt * 16 + m;
#pragma unroll
        for (int r = 0; r < 8; ++r)
            tcol[(size_t)(r + 8 * half) * CH] = acc[r];
    }
}

// ---------------- scatter-add aggregation: agg[col] += t[row] * enorm ----------------
// One wave per edge; each lane owns 4 channels (float4 gather + 4 f32 atomics).
__global__ void k_aggregate(const float* __restrict__ t,
                            const int* __restrict__ row, const int* __restrict__ col,
                            const float* __restrict__ enorm,
                            float* __restrict__ agg, int nE) {
    int e    = (blockIdx.x * blockDim.x + threadIdx.x) >> 5;
    int lane = threadIdx.x & 31;
    if (e >= nE) return;                       // wave-uniform
    int   s = row[e], d = col[e];
    float w = enorm[e];
    float4 v = ((const float4*)(t + (size_t)s * CH))[lane];
    float* dst = agg + (size_t)d * CH + lane * 4;
    atomicAdd(dst + 0, v.x * w);
    atomicAdd(dst + 1, v.y * w);
    atomicAdd(dst + 2, v.z * w);
    atomicAdd(dst + 3, v.w * w);
}

// out = relu(agg + t*inv^2 + b)  -> f16 activations for next GEMM
__global__ void k_finalize_gcn(const float* __restrict__ t, const float* __restrict__ agg,
                               const float* __restrict__ inv, const float* __restrict__ bias,
                               _Float16* __restrict__ hout, int nNodes) {
    int tid = blockIdx.x * blockDim.x + threadIdx.x;       // nNodes*32 threads
    if (tid >= nNodes * 32) return;
    int node = tid >> 5, j = tid & 31;
    float  sn = inv[node]; sn *= sn;
    size_t o  = (size_t)node * CH + j * 4;
    float4 tv = *(const float4*)(t + o);
    float4 av = *(const float4*)(agg + o);
    float4 bv = ((const float4*)bias)[j];
    float r0 = fmaxf(av.x + tv.x * sn + bv.x, 0.0f);
    float r1 = fmaxf(av.y + tv.y * sn + bv.y, 0.0f);
    float r2 = fmaxf(av.z + tv.z * sn + bv.z, 0.0f);
    float r3 = fmaxf(av.w + tv.w * sn + bv.w, 0.0f);
    v4h hv; hv[0] = (_Float16)r0; hv[1] = (_Float16)r1; hv[2] = (_Float16)r2; hv[3] = (_Float16)r3;
    *(v4h*)(hout + o) = hv;
}

// out = relu(t + b)  (fc1)
__global__ void k_finalize_fc(const float* __restrict__ t, const float* __restrict__ bias,
                              _Float16* __restrict__ hout, int nNodes) {
    int tid = blockIdx.x * blockDim.x + threadIdx.x;
    if (tid >= nNodes * 32) return;
    int node = tid >> 5, j = tid & 31;
    size_t o  = (size_t)node * CH + j * 4;
    float4 tv = *(const float4*)(t + o);
    float4 bv = ((const float4*)bias)[j];
    v4h hv;
    hv[0] = (_Float16)fmaxf(tv.x + bv.x, 0.0f);
    hv[1] = (_Float16)fmaxf(tv.y + bv.y, 0.0f);
    hv[2] = (_Float16)fmaxf(tv.z + bv.z, 0.0f);
    hv[3] = (_Float16)fmaxf(tv.w + bv.w, 0.0f);
    *(v4h*)(hout + o) = hv;
}

// fc2 (128 -> 2) + per-graph sum/count pooling. One wave per node.
__global__ void k_fc2_pool(const _Float16* __restrict__ h16, const float* __restrict__ W2,
                           const float* __restrict__ b2, const int* __restrict__ batch,
                           float* __restrict__ gsum, float* __restrict__ gcnt, int nNodes) {
    int node = (blockIdx.x * blockDim.x + threadIdx.x) >> 5;
    int lane = threadIdx.x & 31;
    if (node >= nNodes) return;                // wave-uniform
    v4h hv = *(const v4h*)(h16 + (size_t)node * CH + lane * 4);
    float s0 = 0.0f, s1 = 0.0f;
#pragma unroll
    for (int q = 0; q < 4; ++q) {
        float h = (float)hv[q];
        int   k = lane * 4 + q;
        s0 += h * W2[k * 2 + 0];
        s1 += h * W2[k * 2 + 1];
    }
#pragma unroll
    for (int off = 16; off > 0; off >>= 1) {
        s0 += __shfl_xor(s0, off);
        s1 += __shfl_xor(s1, off);
    }
    if (lane == 0) {
        int g = batch[node];
        atomicAdd(&gsum[g * 2 + 0], s0 + b2[0]);
        atomicAdd(&gsum[g * 2 + 1], s1 + b2[1]);
        atomicAdd(&gcnt[g], 1.0f);
    }
}

__global__ void k_softmax(const float* __restrict__ gsum, const float* __restrict__ gcnt,
                          float* __restrict__ out) {
    int g = threadIdx.x;
    if (g >= N_GRAPHS) return;
    float c  = fmaxf(gcnt[g], 1.0f);
    float a0 = gsum[g * 2 + 0] / c;
    float a1 = gsum[g * 2 + 1] / c;
    float m  = fmaxf(a0, a1);
    float e0 = __expf(a0 - m), e1 = __expf(a1 - m);
    float s  = e0 + e1;
    out[g * 2 + 0] = e0 / s;
    out[g * 2 + 1] = e1 / s;
}

// ---------------- host-side orchestration ----------------

extern "C" void kernel_launch(void* const* d_in, const int* in_sizes, int n_in,
                              void* d_out, int out_size, void* d_ws, size_t ws_size,
                              hipStream_t stream) {
    const float* x    = (const float*)d_in[0];
    const int*   ei   = (const int*)d_in[1];
    const int*   bvec = (const int*)d_in[2];
    const float* W1   = (const float*)d_in[3];
    const float* b1   = (const float*)d_in[4];
    const float* W2   = (const float*)d_in[5];
    const float* b2   = (const float*)d_in[6];
    const float* W3   = (const float*)d_in[7];
    const float* b3   = (const float*)d_in[8];
    const float* fcW1 = (const float*)d_in[9];
    const float* fcb1 = (const float*)d_in[10];
    const float* fcW2 = (const float*)d_in[11];
    const float* fcb2 = (const float*)d_in[12];
    float* out = (float*)d_out;

    char* ws = (char*)d_ws;
    float*    inv   = (float*)(ws + 0);            // N floats (deg -> inv_sqrt in place)
    float*    enorm = (float*)(ws + 200192);       // E floats
    _Float16* h16   = (_Float16*)(ws + 3400192);   // N*128 f16 activations
    float*    t     = (float*)(ws + 16200192);     // N*128 f32 GEMM output
    float*    agg   = (float*)(ws + 41800192);     // N*128 f32 scatter target
    _Float16* wpack = (_Float16*)(ws + 67400192);  // 4 x 16384 f16 packed weights
    float*    gsum  = (float*)(ws + 67531264);     // 128 floats
    float*    gcnt  = (float*)(ws + 67531776);     // 64 floats

    const int* erow = ei;
    const int* ecol = ei + N_EDGES;

    const int BLK = 256;
    const int gE     = (N_EDGES + BLK - 1) / BLK;            // per-edge, 1 thread
    const int gN     = (N_NODES + BLK - 1) / BLK;            // per-node, 1 thread
    const int gEwarp = (N_EDGES * 32 + BLK - 1) / BLK;       // per-edge, 1 wave
    const int gNwarp = (N_NODES * 32 + BLK - 1) / BLK;       // per-node(+chan4), 1 wave
    const int gGemm  = ((N_NODES / 16) * 32 + BLK - 1) / BLK;
    const int gCvt   = (N_NODES * CH / 4 + BLK - 1) / BLK;

    // --- degree / normalization precompute ---
    k_zero<<<gN, BLK, 0, stream>>>(inv, N_NODES);
    k_zero<<<1, BLK, 0, stream>>>(gsum, N_GRAPHS * 2);
    k_zero<<<1, BLK, 0, stream>>>(gcnt, N_GRAPHS);
    k_degree<<<gE, BLK, 0, stream>>>(ecol, inv, N_EDGES);
    k_invsqrt<<<gN, BLK, 0, stream>>>(inv, N_NODES);
    k_edge_norm<<<gE, BLK, 0, stream>>>(erow, ecol, inv, enorm, N_EDGES);

    // --- pack weights into WMMA fragment order ---
    k_pack_w<<<64, BLK, 0, stream>>>(W1,   wpack + 0 * 16384);
    k_pack_w<<<64, BLK, 0, stream>>>(W2,   wpack + 1 * 16384);
    k_pack_w<<<64, BLK, 0, stream>>>(W3,   wpack + 2 * 16384);
    k_pack_w<<<64, BLK, 0, stream>>>(fcW1, wpack + 3 * 16384);

    // --- input features -> f16 ---
    k_cvt_f16<<<gCvt, BLK, 0, stream>>>(x, h16, N_NODES * CH / 4);

    // --- 3 GCN layers ---
    const float* biases[3] = { b1, b2, b3 };
    for (int layer = 0; layer < 3; ++layer) {
        k_gemm_wmma<<<gGemm, BLK, 0, stream>>>(h16, wpack + layer * 16384, t, N_NODES);
        k_zero<<<(N_NODES * CH + BLK - 1) / BLK, BLK, 0, stream>>>(agg, N_NODES * CH);
        k_aggregate<<<gEwarp, BLK, 0, stream>>>(t, erow, ecol, enorm, agg, N_EDGES);
        k_finalize_gcn<<<gNwarp, BLK, 0, stream>>>(t, agg, inv, biases[layer], h16, N_NODES);
    }

    // --- fc1 + relu ---
    k_gemm_wmma<<<gGemm, BLK, 0, stream>>>(h16, wpack + 3 * 16384, t, N_NODES);
    k_finalize_fc<<<gNwarp, BLK, 0, stream>>>(t, fcb1, h16, N_NODES);

    // --- fc2 + mean pool + softmax ---
    k_fc2_pool<<<gNwarp, BLK, 0, stream>>>(h16, fcW2, fcb2, bvec, gsum, gcnt, N_NODES);
    k_softmax<<<1, 64, 0, stream>>>(gsum, gcnt, out);

    (void)in_sizes; (void)n_in; (void)out_size; (void)ws_size;
}
